// Model_37520834298032
// MI455X (gfx1250) — matrix operand order repaired and checked by reading, not compiled, not run on hardware
//
#include <hip/hip_runtime.h>
#include <math.h>

// ---------------- problem constants (from reference) ----------------
#define N_NODES 16384
#define IN_DIM  512
#define HID     256
#define SAMPLE  8
#define INV_TAU 2.0f      // 1/TAU, TAU = 0.5
#define LAMBDA_ 1.0f

// ---------------- optional CDNA5 async global->LDS path ----------------
#if defined(__has_builtin)
#  if __has_builtin(__builtin_amdgcn_global_load_async_to_lds_b128)
#    define USE_ASYNC_LDS 1
#  endif
#  if __has_builtin(__builtin_amdgcn_s_wait_asynccnt)
#    define HAVE_WAIT_ASYNC_BUILTIN 1
#  endif
#  if __has_builtin(__builtin_amdgcn_cvt_pk_bf16_f32)
#    define HAVE_PK_BF16 1
#  endif
#endif
#ifndef USE_ASYNC_LDS
#  define USE_ASYNC_LDS 0
#endif

// builtin prototype (from hipcc diagnostic): (v4i as1*, v4i as3*, Ii, Ii)
typedef int v4i_raw __attribute__((__vector_size__(4 * sizeof(int))));
typedef __attribute__((address_space(1))) v4i_raw glob_v4i;
typedef __attribute__((address_space(3))) v4i_raw lds_v4i;

__device__ __forceinline__ glob_v4i* to_glob_v4(const void* p) {
  return (glob_v4i*)(unsigned long long)p;
}
__device__ __forceinline__ lds_v4i* to_lds_v4(void* p) {
  // generic LDS address: low 32 bits are the LDS offset
  return (lds_v4i*)(unsigned int)(unsigned long long)p;
}
__device__ __forceinline__ void wait_async0() {
#if USE_ASYNC_LDS
#  ifdef HAVE_WAIT_ASYNC_BUILTIN
  __builtin_amdgcn_s_wait_asynccnt(0);
#  else
  asm volatile("s_wait_asynccnt 0x0" ::: "memory");
#  endif
#endif
}

// ---------------- WMMA vector types (gfx1250, wave32) ----------------
typedef __attribute__((ext_vector_type(16))) __bf16 bf16x16;
typedef __attribute__((ext_vector_type(2)))  __bf16 bf16x2;
typedef __attribute__((ext_vector_type(8)))  float  f32x8;

// f32 -> bf16 round-to-nearest-even (finite inputs; no NaN guard in hot path)
__device__ __forceinline__ unsigned short f32_to_bf16_rne(float f) {
  union { float f; unsigned u; } x; x.f = f;
  return (unsigned short)((x.u + 0x7FFFu + ((x.u >> 16) & 1u)) >> 16);
}

// pack two f32 into one dword of two bf16: single v_cvt_pk_bf16_f32
__device__ __forceinline__ unsigned pack2_bf16(float x, float y) {
#ifdef HAVE_PK_BF16
  union { bf16x2 v; unsigned u; } c;
  c.v = __builtin_amdgcn_cvt_pk_bf16_f32(x, y);
  return c.u;
#else
  // builtin not declared for gfx1250, but the ISA instruction exists: emit it directly
  unsigned d;
  asm("v_cvt_pk_bf16_f32 %0, %1, %2" : "=v"(d) : "v"(x), "v"(y));
  return d;
#endif
}

// ---------------- small elementwise kernels ----------------
__global__ void k_f2bf(const float* __restrict__ in, unsigned short* __restrict__ out, long n) {
  long i = (long)blockIdx.x * blockDim.x + threadIdx.x;
  if (i < n) out[i] = f32_to_bf16_rne(in[i]);
}

// out[r][0:cols]=A[r], out[r][cols:2cols]=B[r]  (fp32 -> bf16)
__global__ void k_concat2_bf16(const float* __restrict__ A, const float* __restrict__ B,
                               unsigned short* __restrict__ out, int rows, int cols) {
  long i = (long)blockIdx.x * blockDim.x + threadIdx.x;
  long total = (long)rows * cols * 2;
  if (i >= total) return;
  int w = 2 * cols;
  long r = i / w; int c = (int)(i - r * (long)w);
  float v = (c < cols) ? A[r * cols + c] : B[r * cols + (c - cols)];
  out[i] = f32_to_bf16_rne(v);
}

__global__ void k_bias_cat(const float* __restrict__ a, const float* __restrict__ b,
                           float* __restrict__ out, int h) {
  int i = blockIdx.x * blockDim.x + threadIdx.x;
  if (i < h) out[i] = a[i];
  else if (i < 2 * h) out[i] = b[i - h];
}

__global__ void k_copy_cols(const float* __restrict__ in, int ldin, int coloff,
                            float* __restrict__ out, int ldout, int rows, int cols) {
  long i = (long)blockIdx.x * blockDim.x + threadIdx.x;
  long total = (long)rows * cols;
  if (i >= total) return;
  long r = i / cols; int c = (int)(i - r * (long)cols);
  out[r * ldout + c] = in[r * ldin + coloff + c];
}

// out[c][r] = bf16(in[r][c])   (transpose-convert for Gram B-operands)
__global__ void k_transpose_bf16(const float* __restrict__ in,
                                 unsigned short* __restrict__ out, int rows, int cols) {
  long i = (long)blockIdx.x * blockDim.x + threadIdx.x;
  long total = (long)rows * cols;
  if (i >= total) return;
  long r = i / cols; long c = i - r * (long)cols;
  out[c * rows + r] = f32_to_bf16_rne(in[i]);
}

// wave-per-row L2 normalize (strided in/out)
__global__ void k_l2norm(const float* __restrict__ in, int ldin,
                         float* __restrict__ out, int ldout, int rows, int cols) {
  int row  = blockIdx.x * (blockDim.x >> 5) + (threadIdx.x >> 5);
  int lane = threadIdx.x & 31;
  if (row >= rows) return;
  const float* p = in + (size_t)row * ldin;
  float ss = 0.f;
  for (int c = lane; c < cols; c += 32) { float v = p[c]; ss += v * v; }
  for (int m = 16; m >= 1; m >>= 1) ss += __shfl_xor(ss, m, 32);
  float inv = 1.f / fmaxf(sqrtf(ss), 1e-12f);
  float* q = out + (size_t)row * ldout;
  for (int c = lane; c < cols; c += 32) q[c] = p[c] * inv;
}

// ---------------- shared WMMA mainloop ----------------
// Block = 128 threads = 4 waves; tile BMxBN = 64x64, BK = 32 (one WMMA K-step).
#define BM 64
#define BN 64
#define BK 32

__device__ __forceinline__ void wmma_mainloop(
    const float* __restrict__ A, int lda,
    const unsigned short* __restrict__ B, int ldb,
    int m0, int n0, int K,
    unsigned short (&As)[BM][BK],
#if USE_ASYNC_LDS
    unsigned short (&Bs)[BK][BN],
#else
    unsigned short (&Bs)[BN][BK],
#endif
    f32x8 (&acc)[2][2]) {
  const int tid  = threadIdx.x;
  const int lane = tid & 31;
  const int wave = tid >> 5;
  const int wm = (wave >> 1) * 32;
  const int wn = (wave & 1) * 32;
  const int fm = lane & 15;                // row (A) / col (B) within 16
  const int kA = (lane < 16) ? 0 : 16;     // A-frag K phase (contiguous k = 16h + s)
  const int kB = (lane < 16) ? 0 : 16;     // B-frag K phase

  for (int kb = 0; kb < K; kb += BK) {
    // ---- stage A: 64x32 f32 -> packed bf16 (b128 loads, v_cvt_pk, ds_store_b64) ----
#pragma unroll
    for (int t = 0; t < 4; ++t) {
      int q = tid + t * 128;               // float4 index 0..511
      int r = q >> 3, c4 = (q & 7) * 4;
      const float4 f = *(const float4*)&A[(size_t)(m0 + r) * lda + (kb + c4)];
      uint2 pk;
      pk.x = pack2_bf16(f.x, f.y);
      pk.y = pack2_bf16(f.z, f.w);
      *(uint2*)&As[r][c4] = pk;
    }
    // prefetch the streaming A operand two K-tiles ahead (global_prefetch_b8)
    if (kb + 2 * BK < K) {
      int r = tid >> 3, c4 = (tid & 7) * 4;
      __builtin_prefetch(&A[(size_t)(m0 + r) * lda + (kb + 2 * BK + c4)], 0, 1);
    }
    // ---- stage B: 32x64 bf16 ----
#if USE_ASYNC_LDS
    // direct memory->LDS, 32 lanes x 16B per op, 8 ops per tile (2 per wave)
#pragma unroll
    for (int t = 0; t < 2; ++t) {
      int chunk = wave * 2 + t;            // 0..7
      int r = chunk * 4 + (lane >> 3);     // k row 0..31
      int c8 = (lane & 7) * 8;             // 8 bf16 = 16B
      __builtin_amdgcn_global_load_async_to_lds_b128(
          to_glob_v4(&B[(size_t)(kb + r) * ldb + (n0 + c8)]),
          to_lds_v4(&Bs[r][c8]), 0, 0);
    }
    wait_async0();
#else
#pragma unroll
    for (int t = 0; t < 2; ++t) {
      int q = tid + t * 128;               // 16B chunk 0..255
      int r = q >> 3, c8 = (q & 7) * 8;
      uint4 v = *(const uint4*)&B[(size_t)(kb + r) * ldb + (n0 + c8)];
      const unsigned short* pv = (const unsigned short*)&v;
#pragma unroll
      for (int e = 0; e < 8; e++) Bs[c8 + e][r] = pv[e];
    }
#endif
    __syncthreads();

    union FragA { bf16x16 v; unsigned short u[16]; } a[2];
    union FragB { bf16x16 v; unsigned short u[16]; } b[2];
#pragma unroll
    for (int i = 0; i < 2; i++) {
      int m = wm + i * 16 + fm;
#pragma unroll
      for (int h = 0; h < 8; h++) { a[i].u[h] = As[m][kA + h]; a[i].u[8 + h] = As[m][kA + 8 + h]; }
    }
#pragma unroll
    for (int j = 0; j < 2; j++) {
      int n = wn + j * 16 + fm;
#pragma unroll
      for (int h = 0; h < 16; h++) {
#if USE_ASYNC_LDS
        b[j].u[h] = Bs[kB + h][n];
#else
        b[j].u[h] = Bs[n][kB + h];
#endif
      }
    }
#pragma unroll
    for (int i = 0; i < 2; i++)
#pragma unroll
      for (int j = 0; j < 2; j++)
        acc[i][j] = __builtin_amdgcn_wmma_f32_16x16x32_bf16(
            false, a[i].v, false, b[j].v, (short)0, acc[i][j], false, false);
    __syncthreads();
  }
}

// ---------------- generic WMMA GEMM: C = A(f32) * B(bf16) + bias [relu] ----------------
template <int RELU>
__global__ __launch_bounds__(128)
void k_gemm_wmma(const float* __restrict__ A, int lda,
                 const unsigned short* __restrict__ B, int ldb,
                 const float* __restrict__ bias,   // always valid (zero vector if none)
                 float* __restrict__ C, int ldc, int K) {
  __shared__ unsigned short As[BM][BK];
#if USE_ASYNC_LDS
  __shared__ unsigned short Bs[BK][BN];
#else
  __shared__ unsigned short Bs[BN][BK];
#endif
  const int m0 = (int)blockIdx.y * BM;
  const int n0 = (int)blockIdx.x * BN;

  f32x8 acc[2][2];
#pragma unroll
  for (int i = 0; i < 2; i++)
#pragma unroll
    for (int j = 0; j < 2; j++)
#pragma unroll
      for (int r = 0; r < 8; r++) acc[i][j][r] = 0.f;

  wmma_mainloop(A, lda, B, ldb, m0, n0, K, As, Bs, acc);

  const int lane = threadIdx.x & 31;
  const int wave = threadIdx.x >> 5;
  const int wm = (wave >> 1) * 32;
  const int wn = (wave & 1) * 32;
  const int fm = lane & 15;
  const int mlo = (lane < 16) ? 0 : 8;
#pragma unroll
  for (int i = 0; i < 2; i++)
#pragma unroll
    for (int j = 0; j < 2; j++) {
      int n = n0 + wn + j * 16 + fm;
      float bv = bias[n];
#pragma unroll
      for (int r = 0; r < 8; r++) {
        int m = m0 + wm + i * 16 + mlo + r;
        float v = acc[i][j][r] + bv;
        if (RELU) v = fmaxf(v, 0.f);
        C[(size_t)m * ldc + n] = v;
      }
    }
}

// ---- fused Gram kernel: sums of exp(z_a . z_bT / tau) per row (and optionally per col) ----
__global__ __launch_bounds__(128)
void k_gram_exp_sums(const float* __restrict__ A, int lda,
                     const unsigned short* __restrict__ B, int ldb,
                     float* __restrict__ rowsum, float* __restrict__ colsum, int K) {
  __shared__ unsigned short As[BM][BK];
#if USE_ASYNC_LDS
  __shared__ unsigned short Bs[BK][BN];
#else
  __shared__ unsigned short Bs[BN][BK];
#endif
  const int m0 = (int)blockIdx.y * BM;
  const int n0 = (int)blockIdx.x * BN;

  f32x8 acc[2][2];
#pragma unroll
  for (int i = 0; i < 2; i++)
#pragma unroll
    for (int j = 0; j < 2; j++)
#pragma unroll
      for (int r = 0; r < 8; r++) acc[i][j][r] = 0.f;

  wmma_mainloop(A, lda, B, ldb, m0, n0, K, As, Bs, acc);

  const int lane = threadIdx.x & 31;
  const int wave = threadIdx.x >> 5;
  const int wm = (wave >> 1) * 32;
  const int wn = (wave & 1) * 32;
  const int fm = lane & 15;

  // epilogue: e = exp(c/tau); row sums (lanes sharing a row) + optional col sums
  float colacc0 = 0.f, colacc1 = 0.f;
#pragma unroll
  for (int i = 0; i < 2; i++) {
#pragma unroll
    for (int r = 0; r < 8; r++) {
      float e0 = __expf(acc[i][0][r] * INV_TAU);
      float e1 = __expf(acc[i][1][r] * INV_TAU);
      colacc0 += e0; colacc1 += e1;
      float rs = e0 + e1;
      rs += __shfl_xor(rs, 1, 32);
      rs += __shfl_xor(rs, 2, 32);
      rs += __shfl_xor(rs, 4, 32);
      rs += __shfl_xor(rs, 8, 32);
      if ((lane & 15) == 0) {
        int m = m0 + wm + i * 16 + ((lane < 16) ? 0 : 8) + r;
        atomicAdd(&rowsum[m], rs);
      }
    }
  }
  if (colsum) {
    colacc0 += __shfl_xor(colacc0, 16, 32);
    colacc1 += __shfl_xor(colacc1, 16, 32);
    if (lane < 16) {
      atomicAdd(&colsum[n0 + wn + fm], colacc0);
      atomicAdd(&colsum[n0 + wn + 16 + fm], colacc1);
    }
  }
}

// ---------------- contrastive (gacl) loss: wave per node ----------------
__global__ void k_gacl(const float* __restrict__ projected, const float* __restrict__ u_emd,
                       const float* __restrict__ v_norm, const int* __restrict__ nb_idx,
                       const int* __restrict__ ng_idx, float* __restrict__ gacl_acc, int Nn) {
  int node = blockIdx.x * (blockDim.x >> 5) + (threadIdx.x >> 5);
  int lane = threadIdx.x & 31;
  if (node >= Nn) return;
  const float* pr = projected + (size_t)node * HID;
  const float* vn = v_norm    + (size_t)node * HID;
  float pos[SAMPLE];
  float negsum = 0.f;
  for (int s = 0; s < SAMPLE; s++) {
    const float* u = u_emd + (size_t)nb_idx[node * SAMPLE + s] * HID;
    float d = 0.f;
    for (int c = lane; c < HID; c += 32) d += pr[c] * u[c];
    for (int m = 16; m >= 1; m >>= 1) d += __shfl_xor(d, m, 32);
    pos[s] = __expf(d * INV_TAU);
    const float* w = v_norm + (size_t)ng_idx[node * SAMPLE + s] * HID;
    float dn = 0.f;
    for (int c = lane; c < HID; c += 32) dn += vn[c] * w[c];
    for (int m = 16; m >= 1; m >>= 1) dn += __shfl_xor(dn, m, 32);
    negsum += __expf(dn * INV_TAU);
  }
  if (lane == 0) {
    float neg_score = 0.f, pos_score = 0.f;
    for (int s = 0; s < SAMPLE; s++) { neg_score += logf(pos[s] + negsum); pos_score += logf(pos[s]); }
    atomicAdd(gacl_acc, (LAMBDA_ * neg_score - pos_score) / ((float)Nn * (float)SAMPLE));
  }
}

// ---------------- semi_loss combine: wave per row ----------------
__global__ void k_sem(const float* __restrict__ z1, const float* __restrict__ z2,
                      const float* __restrict__ r1, const float* __restrict__ r2,
                      const float* __restrict__ brow, const float* __restrict__ bcol,
                      float* __restrict__ sem_acc, int Nn) {
  int row  = blockIdx.x * (blockDim.x >> 5) + (threadIdx.x >> 5);
  int lane = threadIdx.x & 31;
  if (row >= Nn) return;
  const float* a = z1 + (size_t)row * HID;
  const float* b = z2 + (size_t)row * HID;
  float d12 = 0.f, d11 = 0.f, d22 = 0.f;
  for (int c = lane; c < HID; c += 32) {
    float x = a[c], y = b[c];
    d12 += x * y; d11 += x * x; d22 += y * y;
  }
  for (int m = 16; m >= 1; m >>= 1) {
    d12 += __shfl_xor(d12, m, 32);
    d11 += __shfl_xor(d11, m, 32);
    d22 += __shfl_xor(d22, m, 32);
  }
  if (lane == 0) {
    float eb = __expf(d12 * INV_TAU);
    float e1 = __expf(d11 * INV_TAU);
    float e2 = __expf(d22 * INV_TAU);
    float l1 = -logf(eb / (r1[row] + brow[row] - e1));
    float l2 = -logf(eb / (r2[row] + bcol[row] - e2));
    atomicAdd(sem_acc, 0.5f * (l1 + l2) / (float)Nn);
  }
}

__global__ void k_final(const float* __restrict__ gacl_acc, const float* __restrict__ sem_acc,
                        float* __restrict__ out) {
  if (threadIdx.x == 0) {
    float g = *gacl_acc, s = *sem_acc;
    out[0] = g + 0.6f * s;   // total
    out[1] = g;              // gacl
    out[2] = s;              // sem
  }
}

// ---------------- host launcher ----------------
extern "C" void kernel_launch(void* const* d_in, const int* in_sizes, int n_in,
                              void* d_out, int out_size, void* d_ws, size_t ws_size,
                              hipStream_t stream) {
  (void)in_sizes; (void)n_in; (void)out_size; (void)ws_size;
  const int N = N_NODES, D = IN_DIM, H = HID;

  const float* adj1 = (const float*)d_in[0];
  const float* adj2 = (const float*)d_in[1];
  const float* x1   = (const float*)d_in[2];
  const float* x2   = (const float*)d_in[3];
  const int*   nb   = (const int*)d_in[4];
  const int*   ng   = (const int*)d_in[5];
  const float* W1   = (const float*)d_in[6];
  const float* b1   = (const float*)d_in[7];
  const float* W2   = (const float*)d_in[8];
  const float* b2   = (const float*)d_in[9];
  const float* tW1  = (const float*)d_in[10];
  const float* tb1  = (const float*)d_in[11];
  const float* tW2  = (const float*)d_in[12];
  const float* tb2  = (const float*)d_in[13];
  const float* pW[3] = {(const float*)d_in[14], (const float*)d_in[16], (const float*)d_in[18]};
  const float* pB[3] = {(const float*)d_in[15], (const float*)d_in[17], (const float*)d_in[19]};
  const float* sW[3] = {(const float*)d_in[20], (const float*)d_in[22], (const float*)d_in[24]};
  const float* sB[3] = {(const float*)d_in[21], (const float*)d_in[23], (const float*)d_in[25]};
  float* out = (float*)d_out;

  // ---- workspace layout ----
  char* base = (char*)d_ws;
  size_t off = 0;
  auto alloc = [&](size_t bytes) -> void* {
    void* p = base + off;
    off += (bytes + 255) & ~(size_t)255;
    return p;
  };
  const size_t NH = (size_t)N * H;

  // accumulators + zero-bias first (zeroed in one memset)
  float* r1   = (float*)alloc((size_t)N * 4);
  float* r2   = (float*)alloc((size_t)N * 4);
  float* brow = (float*)alloc((size_t)N * 4);
  float* bcol = (float*)alloc((size_t)N * 4);
  float* gacl_acc = (float*)alloc(256);
  float* sem_acc  = (float*)alloc(256);
  float* zbias    = (float*)alloc(2 * H * 4);   // stays zero: "no bias"
  size_t zero_bytes = off;

  unsigned short* W1b  = (unsigned short*)alloc((size_t)D * H * 2);
  unsigned short* tW1b = (unsigned short*)alloc((size_t)D * H * 2);
  unsigned short* W2b  = (unsigned short*)alloc((size_t)H * H * 2);
  unsigned short* tW2b = (unsigned short*)alloc((size_t)H * H * 2);
  unsigned short* pWb[3], *sWb[3];
  for (int i = 0; i < 3; i++) pWb[i] = (unsigned short*)alloc((size_t)H * H * 2);
  for (int i = 0; i < 3; i++) sWb[i] = (unsigned short*)alloc((size_t)H * H * 2);
  float* bc1 = (float*)alloc(2 * H * 4);
  float* bc2 = (float*)alloc(2 * H * 4);
  float* bc3 = (float*)alloc(2 * H * 4);

  float* P1  = (float*)alloc(NH * 4);
  float* P2  = (float*)alloc(NH * 4);
  float* Pt1 = (float*)alloc(NH * 4);
  unsigned short* Bcat1 = (unsigned short*)alloc(NH * 2 * 2);   // N x 512 bf16
  unsigned short* Bcat2 = (unsigned short*)alloc(NH * 2 * 2);
  float* C1 = (float*)alloc(NH * 2 * 4);                        // [h1 | h2]
  float* C2 = (float*)alloc(NH * 2 * 4);                        // [inner | innert]
  float* C3 = (float*)alloc(NH * 2 * 4);                        // [h3 | u_pre]
  float* u_emd = (float*)alloc(NH * 4);
  float* T0 = (float*)alloc(NH * 4);
  float* T1 = (float*)alloc(NH * 4);
  float* projected = (float*)alloc(NH * 4);
  float* h1p = (float*)alloc(NH * 4);
  float* h2p = (float*)alloc(NH * 4);
  float* vnorm = (float*)alloc(NH * 4);
  unsigned short* z1t = (unsigned short*)alloc(NH * 2);
  unsigned short* z2t = (unsigned short*)alloc(NH * 2);
  // reuse dead regions
  float* mid  = P1;                 // P1/P2 dead after Bcat1/Bcat2 built
  float* midt = P2;
  unsigned short* Bcat3 = Bcat1;    // Bcat1 dead after C1 GEMM

  (void)hipMemsetAsync(d_ws, 0, zero_bytes, stream);

  const int T = 256;
  auto cdiv = [](long a, long b) { return (int)((a + b - 1) / b); };
  dim3 blk128(128);
  auto gemm = [&](const float* A, int lda, const unsigned short* B, int ldb,
                  const float* bias, float* C, int ldc, int M, int Nc, int K, int relu) {
    dim3 grid(Nc / BN, M / BM);
    if (relu)
      k_gemm_wmma<1><<<grid, blk128, 0, stream>>>(A, lda, B, ldb, bias, C, ldc, K);
    else
      k_gemm_wmma<0><<<grid, blk128, 0, stream>>>(A, lda, B, ldb, bias, C, ldc, K);
  };

  // weights -> bf16, bias concats
  k_f2bf<<<cdiv((long)D * H, T), T, 0, stream>>>(W1,  W1b,  (long)D * H);
  k_f2bf<<<cdiv((long)D * H, T), T, 0, stream>>>(tW1, tW1b, (long)D * H);
  k_f2bf<<<cdiv((long)H * H, T), T, 0, stream>>>(W2,  W2b,  (long)H * H);
  k_f2bf<<<cdiv((long)H * H, T), T, 0, stream>>>(tW2, tW2b, (long)H * H);
  for (int i = 0; i < 3; i++) {
    k_f2bf<<<cdiv((long)H * H, T), T, 0, stream>>>(pW[i], pWb[i], (long)H * H);
    k_f2bf<<<cdiv((long)H * H, T), T, 0, stream>>>(sW[i], sWb[i], (long)H * H);
  }
  k_bias_cat<<<cdiv(2 * H, T), T, 0, stream>>>(b1, b1,  bc1, H);
  k_bias_cat<<<cdiv(2 * H, T), T, 0, stream>>>(b1, tb1, bc2, H);
  k_bias_cat<<<cdiv(2 * H, T), T, 0, stream>>>(b2, tb2, bc3, H);

  // feature projections: P = x @ W  (M=N, K=512, Nc=256)
  gemm(x1, D, W1b,  H, zbias, P1,  H, N, H, D, 0);
  gemm(x2, D, W1b,  H, zbias, P2,  H, N, H, D, 0);
  gemm(x1, D, tW1b, H, zbias, Pt1, H, N, H, D, 0);
  k_concat2_bf16<<<cdiv((long)N * 2 * H, T), T, 0, stream>>>(P1, P2,  Bcat1, N, H);
  k_concat2_bf16<<<cdiv((long)N * 2 * H, T), T, 0, stream>>>(P1, Pt1, Bcat2, N, H);

  // fused adjacency passes (512-wide): adj1 read once, adj2 twice
  gemm(adj1, N, Bcat1, 2 * H, bc1, C1, 2 * H, N, 2 * H, N, 0);   // [h1|h2]
  gemm(adj2, N, Bcat2, 2 * H, bc2, C2, 2 * H, N, 2 * H, N, 0);   // [inner|innert]

  gemm(C2,     2 * H, W2b,  H, zbias, mid,  H, N, H, H, 0);      // inner @ W2
  gemm(C2 + H, 2 * H, tW2b, H, zbias, midt, H, N, H, H, 0);      // innert @ tW2
  k_concat2_bf16<<<cdiv((long)N * 2 * H, T), T, 0, stream>>>(mid, midt, Bcat3, N, H);
  gemm(adj2, N, Bcat3, 2 * H, bc3, C3, 2 * H, N, 2 * H, N, 0);   // [h3|u_pre]

  // h3 output + normalizations
  k_copy_cols<<<cdiv((long)N * H, T), T, 0, stream>>>(C3, 2 * H, 0, out + 3, H, N, H);
  dim3 rowsGrid(N / 8);
  k_l2norm<<<rowsGrid, 256, 0, stream>>>(C3 + H, 2 * H, u_emd, H, N, H);   // u_emd
  k_l2norm<<<rowsGrid, 256, 0, stream>>>(C3,     2 * H, vnorm, H, N, H);   // v_norm

  // projected = l2norm(mlp(h3, pW, pB))
  gemm(C3, 2 * H, pWb[0], H, pB[0], T0, H, N, H, H, 1);
  gemm(T0, H,     pWb[1], H, pB[1], T1, H, N, H, H, 1);
  gemm(T1, H,     pWb[2], H, pB[2], T0, H, N, H, H, 0);
  k_l2norm<<<rowsGrid, 256, 0, stream>>>(T0, H, projected, H, N, H);

  // h1_proj / h2_proj = l2norm(mlp(h1/h2, sW, sB))
  gemm(C1, 2 * H, sWb[0], H, sB[0], T0, H, N, H, H, 1);
  gemm(T0, H,     sWb[1], H, sB[1], T1, H, N, H, H, 1);
  gemm(T1, H,     sWb[2], H, sB[2], T0, H, N, H, H, 0);
  k_l2norm<<<rowsGrid, 256, 0, stream>>>(T0, H, h1p, H, N, H);

  gemm(C1 + H, 2 * H, sWb[0], H, sB[0], T0, H, N, H, H, 1);
  gemm(T0, H,         sWb[1], H, sB[1], T1, H, N, H, H, 1);
  gemm(T1, H,         sWb[2], H, sB[2], T0, H, N, H, H, 0);
  k_l2norm<<<rowsGrid, 256, 0, stream>>>(T0, H, h2p, H, N, H);

  // contrastive loss
  k_gacl<<<rowsGrid, 256, 0, stream>>>(projected, u_emd, vnorm, nb, ng, gacl_acc, N);

  // semi loss: 3 fused Gram passes (z2@z1^T row-sums == z1@z2^T col-sums)
  k_transpose_bf16<<<cdiv((long)N * H, T), T, 0, stream>>>(h1p, z1t, N, H);
  k_transpose_bf16<<<cdiv((long)N * H, T), T, 0, stream>>>(h2p, z2t, N, H);
  dim3 gGrid(N / BN, N / BM);
  k_gram_exp_sums<<<gGrid, blk128, 0, stream>>>(h1p, H, z1t, N, r1,   nullptr, H);
  k_gram_exp_sums<<<gGrid, blk128, 0, stream>>>(h2p, H, z2t, N, r2,   nullptr, H);
  k_gram_exp_sums<<<gGrid, blk128, 0, stream>>>(h1p, H, z2t, N, brow, bcol,    H);
  k_sem<<<rowsGrid, 256, 0, stream>>>(h1p, h2p, r1, r2, brow, bcol, sem_acc, N);

  k_final<<<1, 32, 0, stream>>>(gacl_acc, sem_acc, out);
}
